// TransformerEncoder_45380624449578
// MI455X (gfx1250) — compile-verified, matrix operand order checked by reading, not run on hardware
//
#include <hip/hip_runtime.h>
#include <hip/hip_bf16.h>

// ---------------------------------------------------------------------------
// Types for WMMA fragments (CDNA5 wave32, v_wmma_f32_16x16x32_f16)
// ---------------------------------------------------------------------------
typedef _Float16 v16h __attribute__((ext_vector_type(16)));
typedef _Float16 v8h  __attribute__((ext_vector_type(8)));
typedef _Float16 v4h  __attribute__((ext_vector_type(4)));
typedef float    v8f  __attribute__((ext_vector_type(8)));
typedef unsigned int v4u __attribute__((ext_vector_type(4)));
typedef int      v8i  __attribute__((ext_vector_type(8)));
typedef int      v4i  __attribute__((ext_vector_type(4)));

union H16 { v16h v; v8h h[2]; };

__device__ __forceinline__ v8f wmma32(v16h a, v16h b, v8f c) {
  // D = A(16x32 f16) * B(32x16 f16) + C(16x16 f32)
  return __builtin_amdgcn_wmma_f32_16x16x32_f16(
      false, a, false, b, (short)0, c, false, false);
}

// convert 8 contiguous f32 (LDS) -> 8 f16 fragment elements
__device__ __forceinline__ v8h cvt8h(const float* __restrict__ p) {
  v8h r;
#pragma unroll
  for (int i = 0; i < 8; ++i) r[i] = (_Float16)p[i];
  return r;
}

#define S_LEN  1024
#define DMODEL 512
#define NHEAD  8
#define HDIM   64
#define FDIM   2048
#define NLAYER 6

// ---------------------------------------------------------------------------
// Tensor Data Mover: 2D tile global->LDS (raw bytes, f32), ISA 08 section 8.
// Descriptor: group0 = {count, lds_addr, global_addr, type=2};
// group1 = {data_size=4B, tensor dims, tile dims, dim0 stride}.
// ---------------------------------------------------------------------------
#if defined(__AMDGCN__) && __has_builtin(__builtin_amdgcn_tensor_load_to_lds)
#define USE_TDM 1
#else
#define USE_TDM 0
#endif

#if USE_TDM
__device__ __forceinline__ void tdm_load_2d(
    unsigned lds_off, const void* gaddr,
    unsigned tensor_d0, unsigned tensor_d1,
    unsigned tile_d0, unsigned tile_d1,
    unsigned long long stride0 /* elements */)
{
  unsigned long long ga = (unsigned long long)gaddr;
  v4u g0;
  g0[0] = 1u;                                            // count=1 (valid), user mode
  g0[1] = lds_off;                                       // lds_addr (bytes)
  g0[2] = (unsigned)(ga & 0xffffffffu);                  // global_addr[31:0]
  g0[3] = (unsigned)((ga >> 32) & 0x01ffffffu) | (2u << 30); // ga[56:32] | type=2
  v8i g1;
  g1[0] = (int)(2u << 16);                               // data_size=2 -> 4 bytes
  g1[1] = (int)((tensor_d0 & 0xffffu) << 16);            // tensor_dim0[15:0]
  g1[2] = (int)((tensor_d0 >> 16) | ((tensor_d1 & 0xffffu) << 16));
  g1[3] = (int)((tensor_d1 >> 16) | (tile_d0 << 16));    // tile_dim0
  g1[4] = (int)(tile_d1 & 0xffffu);                      // tile_dim1 (tile_dim2=0)
  g1[5] = (int)(stride0 & 0xffffffffu);                  // tensor_dim0_stride[31:0]
  g1[6] = (int)((stride0 >> 32) & 0xffffu);              // stride[47:32]
  g1[7] = 0;
  v4i g2 = {0, 0, 0, 0};
  v4i g3 = {0, 0, 0, 0};
#if __clang_major__ >= 23
  v8i g4 = {0, 0, 0, 0, 0, 0, 0, 0};
  __builtin_amdgcn_tensor_load_to_lds(g0, g1, g2, g3, g4, 0);
#else
  __builtin_amdgcn_tensor_load_to_lds(g0, g1, g2, g3, 0);
#endif
}

__device__ __forceinline__ void tdm_wait0() {
#if __has_builtin(__builtin_amdgcn_s_wait_tensorcnt)
  __builtin_amdgcn_s_wait_tensorcnt(0);
#else
  asm volatile("s_wait_tensorcnt 0x0" ::: "memory");
#endif
}
#endif // USE_TDM

// ---------------------------------------------------------------------------
// Tiled WMMA GEMM: out[M,N] = epilogue( A[M,K] @ W[K,N] )
// Tile 128x128, BK=32, 256 threads = 8 waves; wave tile 64x32 = 4x2 frags.
// A tile staged raw f32 by the TDM (wave 0 issues; others stage W); W tile
// converted f16 + transposed by VALU. f32->f16 for A happens at frag build.
// EPI: 0=none 1=+bias 2=+bias+res 3=silu(acc)*extra 4=+res 5=+bias+sinusoidalPE
// ---------------------------------------------------------------------------
#define GT_M 128
#define GT_N 128
#define GT_K 32
#define GST  40   // padded f16 LDS row stride in halfs (80B, 16B multiple)

template<int EPI>
__global__ __launch_bounds__(256) void gemm_k(
    const float* __restrict__ A, const float* __restrict__ W,
    const float* __restrict__ bias, const float* __restrict__ extra,
    float* __restrict__ out, int M, int N, int K)
{
  __shared__ __align__(16) float    sAF32[GT_M * GT_K];  // [row][k] raw f32
  __shared__ __align__(16) _Float16 sB[GT_N * GST];      // transposed: [n][k]

  const int tid  = threadIdx.x;
  const int lane = tid & 31;
  const int wave = tid >> 5;
  const int lrow = lane & 15;
  const int lgrp = lane >> 4;
  const int wm   = wave >> 2;   // 0..1 -> 64-row slice
  const int wn   = wave & 3;    // 0..3 -> 32-col slice
  const int rowBase = blockIdx.y * GT_M;
  const int colBase = blockIdx.x * GT_N;
#if USE_TDM
  const unsigned sA_off = (unsigned)(unsigned long long)(const void*)sAF32;
#endif

  v8f acc[4][2];
  v8f z = {};
  for (int i = 0; i < 4; ++i) for (int j = 0; j < 2; ++j) acc[i][j] = z;

  for (int k0 = 0; k0 < K; k0 += GT_K) {
    // ---- stage A tile: 128x32 f32, DMA'd by the Tensor Data Mover ----
#if USE_TDM
    if (wave == 0)
      tdm_load_2d(sA_off, &A[(size_t)rowBase * K + k0],
                  (unsigned)K, 1u << 20, GT_K, GT_M, (unsigned long long)K);
#else
    for (int it = 0; it < 4; ++it) {
      int idx = tid + it * 256;            // 1024 float4 slots
      int r = idx >> 3, q = (idx & 7) * 4;
      const float4 f = *reinterpret_cast<const float4*>(
          &A[(size_t)(rowBase + r) * K + k0 + q]);
      *reinterpret_cast<float4*>(&sAF32[r * GT_K + q]) = f;
    }
#endif
    // ---- stage W tile: 32x128 f32 -> transposed f16 LDS [n][k] ----
    for (int it = 0; it < 4; ++it) {
      int idx = tid + it * 256;
      int kr = idx >> 5, nc = (idx & 31) * 4;
      const float4 f = *reinterpret_cast<const float4*>(
          &W[(size_t)(k0 + kr) * N + colBase + nc]);
      sB[(nc + 0) * GST + kr] = (_Float16)f.x;
      sB[(nc + 1) * GST + kr] = (_Float16)f.y;
      sB[(nc + 2) * GST + kr] = (_Float16)f.z;
      sB[(nc + 3) * GST + kr] = (_Float16)f.w;
    }
    if (k0 + GT_K < K)
      __builtin_prefetch(&W[(size_t)(k0 + GT_K + (tid >> 5)) * N + colBase], 0, 1);
#if USE_TDM
    if (wave == 0) tdm_wait0();
#endif
    __syncthreads();

    // ---- fragments + WMMA ----
    H16 bf[2];
    for (int fj = 0; fj < 2; ++fj) {
      int n = wn * 32 + fj * 16 + lrow;
      bf[fj].h[0] = *reinterpret_cast<const v8h*>(&sB[n * GST + 8 * lgrp]);
      bf[fj].h[1] = *reinterpret_cast<const v8h*>(&sB[n * GST + 16 + 8 * lgrp]);
    }
    for (int fi = 0; fi < 4; ++fi) {
      int r = wm * 64 + fi * 16 + lrow;
      const float* arow = &sAF32[r * GT_K];
      H16 af;
      af.h[0] = cvt8h(arow + 8 * lgrp);         // k = 8g..8g+7
      af.h[1] = cvt8h(arow + 16 + 8 * lgrp);    // k = 16+8g..16+8g+7
      acc[fi][0] = wmma32(af.v, bf[0].v, acc[fi][0]);
      acc[fi][1] = wmma32(af.v, bf[1].v, acc[fi][1]);
    }
    __syncthreads();
  }

  // ---- epilogue: C layout row = r + 8*(lane/16), col = lane%16 ----
  for (int fi = 0; fi < 4; ++fi)
    for (int fj = 0; fj < 2; ++fj) {
      int col = colBase + wn * 32 + fj * 16 + lrow;
      for (int r = 0; r < 8; ++r) {
        int row = rowBase + wm * 64 + fi * 16 + r + 8 * lgrp;
        size_t idx = (size_t)row * N + col;
        float v = acc[fi][fj][r];
        if (EPI == 1) {
          v += bias[col];
        } else if (EPI == 2) {
          v += bias[col] + extra[idx];
        } else if (EPI == 3) {
          v = (v / (1.f + expf(-v))) * extra[idx];   // silu(G) * U
        } else if (EPI == 4) {
          v += extra[idx];
        } else if (EPI == 5) {
          v += bias[col];
          int s = row & (S_LEN - 1);                 // M = B*S, s = row % S
          float fr  = expf(-9.210340371976184f * (float)(col & ~1) / (float)N);
          float ang = (float)s * fr;
          v += (col & 1) ? cosf(ang) : sinf(ang);
        }
        out[idx] = v;
      }
    }
}

// ---------------------------------------------------------------------------
// Flash attention: one block per (b, h, 64-row q block); 4 waves x 16 q rows.
// K tile DMA'd raw f32 by the TDM while waves transpose-convert the V tile;
// QK^T and PV on WMMA; online softmax (running row max/sum) in LDS per wave.
// ---------------------------------------------------------------------------
#define ATT_KT 64
#define PST    72   // padded f16 LDS row stride in halfs (144B, 16B multiple)

__global__ __launch_bounds__(128) void attn_k(
    const float* __restrict__ Q, const float* __restrict__ K,
    const float* __restrict__ V, float* __restrict__ O)
{
  __shared__ __align__(16) float    sKf32[ATT_KT * HDIM]; // [kpos][kdim] raw f32
  __shared__ __align__(16) _Float16 sVT[HDIM * PST];      // [hd][kpos] f16
  __shared__ __align__(16) float    scb[4][16][ATT_KT];   // scores per wave
  __shared__ __align__(16) _Float16 pb[4][16 * PST];      // exp(P) per wave, f16
  __shared__ float rowm[4][16], rowl[4][16], rsc[4][16];

  const int tid  = threadIdx.x;
  const int lane = tid & 31;
  const int wave = tid >> 5;
  const int lrow = lane & 15;
  const int lgrp = lane >> 4;
  const int qblk = blockIdx.x & 15;          // S/64 = 16
  const int bh   = blockIdx.x >> 4;
  const int h    = bh & (NHEAD - 1);
  const int b    = bh >> 3;

  const float* Qp = Q + ((size_t)b * S_LEN) * DMODEL + h * HDIM;
  const float* Kp = K + ((size_t)b * S_LEN) * DMODEL + h * HDIM;
  const float* Vp = V + ((size_t)b * S_LEN) * DMODEL + h * HDIM;
#if USE_TDM
  const unsigned sK_off = (unsigned)(unsigned long long)(const void*)sKf32;
#endif

  if (lane < 16) { rowm[wave][lane] = -1e30f; rowl[wave][lane] = 0.f; }

  // Q fragments for this wave's 16 rows (kdim 0..31 and 32..63), kept in VGPRs
  H16 qa[2];
  {
    int qrow = qblk * 64 + wave * 16 + lrow;
    for (int f = 0; f < 2; ++f)
      for (int j = 0; j < 16; ++j) {
        int kd = f * 32 + (j & 7) + 8 * lgrp + 16 * (j >> 3);
        qa[f].v[j] = (_Float16)Qp[(size_t)qrow * DMODEL + kd];
      }
  }
  v8f z = {};
  v8f ctxa[4];
  for (int i = 0; i < 4; ++i) ctxa[i] = z;
  __syncthreads();

  for (int kt = 0; kt < S_LEN / ATT_KT; ++kt) {
    const int kbase = kt * ATT_KT;
    // ---- K tile via TDM (issue first so DMA overlaps V staging) ----
#if USE_TDM
    if (wave == 0)
      tdm_load_2d(sK_off, &Kp[(size_t)kbase * DMODEL],
                  DMODEL, 1u << 20, HDIM, ATT_KT, DMODEL);
#endif
    // ---- V tile: f32 -> f16, transposed [hd][kpos] ----
    for (int it = 0; it < 8; ++it) {
      int idx = tid + it * 128;            // 1024 float4 = 64 rows x 16 quads
      int r = idx >> 4, q = (idx & 15) * 4;
      const float4 fv = *reinterpret_cast<const float4*>(
          &Vp[(size_t)(kbase + r) * DMODEL + q]);
      sVT[(q + 0) * PST + r] = (_Float16)fv.x;
      sVT[(q + 1) * PST + r] = (_Float16)fv.y;
      sVT[(q + 2) * PST + r] = (_Float16)fv.z;
      sVT[(q + 3) * PST + r] = (_Float16)fv.w;
#if !USE_TDM
      const float4 fk = *reinterpret_cast<const float4*>(
          &Kp[(size_t)(kbase + r) * DMODEL + q]);
      *reinterpret_cast<float4*>(&sKf32[r * HDIM + q]) = fk;
#endif
    }
#if USE_TDM
    if (wave == 0) tdm_wait0();
#endif
    __syncthreads();

    // ---- scores: S = Q(16x64) * K^T(64x64tile), 2 WMMA per 16-col tile ----
    for (int ct = 0; ct < 4; ++ct) {
      int n = ct * 16 + lrow;
      const float* krow = &sKf32[n * HDIM];
      H16 kb0, kb1;
      kb0.h[0] = cvt8h(krow + 8 * lgrp);
      kb0.h[1] = cvt8h(krow + 16 + 8 * lgrp);
      kb1.h[0] = cvt8h(krow + 32 + 8 * lgrp);
      kb1.h[1] = cvt8h(krow + 48 + 8 * lgrp);
      v8f sf = z;
      sf = wmma32(qa[0].v, kb0.v, sf);
      sf = wmma32(qa[1].v, kb1.v, sf);
      for (int r = 0; r < 8; ++r)
        scb[wave][r + 8 * lgrp][ct * 16 + lrow] = sf[r] * 0.125f; // 1/sqrt(64)
    }
    __builtin_amdgcn_wave_barrier();

    // ---- online softmax per row (lanes 0..15 each own one row) ----
    if (lane < 16) {
      float mo = rowm[wave][lane];
      float mt = mo;
      for (int c = 0; c < ATT_KT; ++c) mt = fmaxf(mt, scb[wave][lane][c]);
      float scale = expf(mo - mt);
      float ls = rowl[wave][lane] * scale;
      for (int c = 0; c < ATT_KT; ++c) {
        float e = expf(scb[wave][lane][c] - mt);
        pb[wave][lane * PST + c] = (_Float16)e;
        ls += e;
      }
      rowm[wave][lane] = mt;
      rowl[wave][lane] = ls;
      rsc[wave][lane]  = scale;
    }
    __builtin_amdgcn_wave_barrier();

    // ---- rescale running context by exp(m_old - m_new) ----
    float s8[8];
    for (int r = 0; r < 8; ++r) s8[r] = rsc[wave][r + 8 * lgrp];
    for (int hf = 0; hf < 4; ++hf)
      for (int r = 0; r < 8; ++r) ctxa[hf][r] *= s8[r];

    // ---- ctx += P(16x64) * V(64x64): 2 k-steps x 4 hd fragments ----
    for (int ks = 0; ks < 2; ++ks) {
      H16 pa;
      pa.h[0] = *reinterpret_cast<const v8h*>(
          &pb[wave][lrow * PST + ks * 32 + 8 * lgrp]);
      pa.h[1] = *reinterpret_cast<const v8h*>(
          &pb[wave][lrow * PST + ks * 32 + 16 + 8 * lgrp]);
      for (int hf = 0; hf < 4; ++hf) {
        int hd = hf * 16 + lrow;
        H16 vbf;
        vbf.h[0] = *reinterpret_cast<const v8h*>(
            &sVT[hd * PST + ks * 32 + 8 * lgrp]);
        vbf.h[1] = *reinterpret_cast<const v8h*>(
            &sVT[hd * PST + ks * 32 + 16 + 8 * lgrp]);
        ctxa[hf] = wmma32(pa.v, vbf.v, ctxa[hf]);
      }
    }
    __syncthreads();
  }

  // ---- normalize and write ctx ----
  float l8[8];
  for (int r = 0; r < 8; ++r) l8[r] = rowl[wave][r + 8 * lgrp];
  for (int hf = 0; hf < 4; ++hf)
    for (int r = 0; r < 8; ++r) {
      int srow = qblk * 64 + wave * 16 + r + 8 * lgrp;
      O[((size_t)b * S_LEN + srow) * DMODEL + h * HDIM + hf * 16 + lrow] =
          ctxa[hf][r] / l8[r];
    }
}

// ---------------------------------------------------------------------------
// RMSNorm over D=512 (one block per token row)
// ---------------------------------------------------------------------------
__global__ __launch_bounds__(256) void rmsnorm_k(
    const float* __restrict__ x, const float* __restrict__ g,
    float* __restrict__ o)
{
  __shared__ float red[256];
  int row = blockIdx.x, tid = threadIdx.x;
  const float* xr = x + (size_t)row * DMODEL;
  float v0 = xr[tid], v1 = xr[tid + 256];
  red[tid] = v0 * v0 + v1 * v1;
  __syncthreads();
  for (int s = 128; s > 0; s >>= 1) {
    if (tid < s) red[tid] += red[tid + s];
    __syncthreads();
  }
  float inv = rsqrtf(red[0] / (float)DMODEL + 1e-5f);
  float* orow = o + (size_t)row * DMODEL;
  orow[tid]       = v0 * g[tid] * inv;
  orow[tid + 256] = v1 * g[tid + 256] * inv;
}

// ---------------------------------------------------------------------------
// LayerGate: g[b] = (sigmoid(mean_s(x[b]) . gw + gb) > 0.5) ? 1 : 0
// ---------------------------------------------------------------------------
__global__ __launch_bounds__(256) void gate_k(
    const float* __restrict__ x, const float* __restrict__ gw,
    const float* __restrict__ gb, float* __restrict__ g)
{
  __shared__ float red[256];
  int b = blockIdx.x, tid = threadIdx.x;
  const float* xp = x + (size_t)b * S_LEN * DMODEL;
  float acc = 0.f;
  for (int i = tid; i < S_LEN * DMODEL; i += 256)
    acc += xp[i] * gw[i & (DMODEL - 1)];
  red[tid] = acc;
  __syncthreads();
  for (int s = 128; s > 0; s >>= 1) {
    if (tid < s) red[tid] += red[tid + s];
    __syncthreads();
  }
  if (tid == 0) {
    float p = 1.f / (1.f + expf(-(red[0] / (float)S_LEN + gb[0])));
    g[b] = (p > 0.5f) ? 1.f : 0.f;
  }
}

__global__ __launch_bounds__(256) void blend_k(
    float* __restrict__ x, const float* __restrict__ y,
    const float* __restrict__ g)
{
  size_t i = (size_t)blockIdx.x * 256 + threadIdx.x;
  int b = (int)(i >> 19);                 // S*D = 2^19
  float gv = g[b];
  x[i] = gv * y[i] + (1.f - gv) * x[i];
}

// ---------------------------------------------------------------------------
// Host orchestration
// ---------------------------------------------------------------------------
extern "C" void kernel_launch(void* const* d_in, const int* in_sizes, int n_in,
                              void* d_out, int out_size, void* d_ws, size_t ws_size,
                              hipStream_t stream)
{
  (void)in_sizes; (void)n_in; (void)out_size; (void)ws_size;
  const float* src  = (const float*)d_in[0];
  const float* in_w = (const float*)d_in[1];
  const float* in_b = (const float*)d_in[2];
  const float* Wq   = (const float*)d_in[3];
  const float* Wk   = (const float*)d_in[4];
  const float* Wv   = (const float*)d_in[5];
  const float* Wo   = (const float*)d_in[6];
  const float* bq   = (const float*)d_in[7];
  const float* bk   = (const float*)d_in[8];
  const float* bv   = (const float*)d_in[9];
  const float* bo   = (const float*)d_in[10];
  const float* n1   = (const float*)d_in[11];
  const float* n2   = (const float*)d_in[12];
  const float* Wg   = (const float*)d_in[13];
  const float* Wu   = (const float*)d_in[14];
  const float* Wd   = (const float*)d_in[15];
  const float* gw   = (const float*)d_in[16];
  const float* gb   = (const float*)d_in[17];
  const float* fs   = (const float*)d_in[18];
  float* out = (float*)d_out;

  const size_t NX = (size_t)8 * S_LEN * DMODEL;   // 4,194,304 floats
  float* xb = (float*)d_ws;
  float* yb = xb + NX;
  float* hb = yb + NX;
  float* qb = hb + NX;
  float* kb = qb + NX;
  float* vb = kb + NX;
  float* cb = vb + NX;
  float* ub = cb + NX;                            // B*S*F floats
  float* gv = ub + (size_t)8 * S_LEN * FDIM;

  const int M = 8 * S_LEN;                        // 8192 token rows
  dim3 blk(256);
  dim3 gD(DMODEL / GT_N, M / GT_M);               // (4, 64)
  dim3 gF(FDIM / GT_N, M / GT_M);                 // (16, 64)

  // input adapter + sinusoidal positional encoding (fused epilogue)
  gemm_k<5><<<gD, blk, 0, stream>>>(src, in_w, in_b, nullptr, xb, M, DMODEL, 64);

  for (int l = 0; l < NLAYER; ++l) {
    const float* Wq_l = Wq + (size_t)l * DMODEL * DMODEL;
    const float* Wk_l = Wk + (size_t)l * DMODEL * DMODEL;
    const float* Wv_l = Wv + (size_t)l * DMODEL * DMODEL;
    const float* Wo_l = Wo + (size_t)l * DMODEL * DMODEL;
    const float* Wg_l = Wg + (size_t)l * DMODEL * FDIM;
    const float* Wu_l = Wu + (size_t)l * DMODEL * FDIM;
    const float* Wd_l = Wd + (size_t)l * FDIM * DMODEL;

    // pre-norm attention
    rmsnorm_k<<<M, 256, 0, stream>>>(xb, n1 + l * DMODEL, hb);
    gemm_k<1><<<gD, blk, 0, stream>>>(hb, Wq_l, bq + l * DMODEL, nullptr, qb, M, DMODEL, DMODEL);
    gemm_k<1><<<gD, blk, 0, stream>>>(hb, Wk_l, bk + l * DMODEL, nullptr, kb, M, DMODEL, DMODEL);
    gemm_k<1><<<gD, blk, 0, stream>>>(hb, Wv_l, bv + l * DMODEL, nullptr, vb, M, DMODEL, DMODEL);
    attn_k<<<8 * NHEAD * (S_LEN / 64), 128, 0, stream>>>(qb, kb, vb, cb);
    gemm_k<2><<<gD, blk, 0, stream>>>(cb, Wo_l, bo + l * DMODEL, xb, yb, M, DMODEL, DMODEL);

    // pre-norm SwiGLU FFN (G-GEMM fuses silu(G)*U in place into ub)
    rmsnorm_k<<<M, 256, 0, stream>>>(yb, n2 + l * DMODEL, hb);
    gemm_k<0><<<gF, blk, 0, stream>>>(hb, Wu_l, nullptr, nullptr, ub, M, FDIM, DMODEL);
    gemm_k<3><<<gF, blk, 0, stream>>>(hb, Wg_l, nullptr, ub, ub, M, FDIM, DMODEL);
    gemm_k<4><<<gD, blk, 0, stream>>>(ub, Wd_l, nullptr, yb, yb, M, DMODEL, FDIM);

    // LayerGate (reads layer input xb), then blend into xb
    gate_k<<<8, 256, 0, stream>>>(xb, gw + l * DMODEL, gb + l, gv);
    blend_k<<<(int)(NX / 256), 256, 0, stream>>>(xb, yb, gv);
  }

  rmsnorm_k<<<M, 256, 0, stream>>>(xb, fs, out);
}
